// CustomLSTMLayer_61237643706592
// MI455X (gfx1250) — matrix operand order repaired and checked by reading, not compiled, not run on hardware
//
#include <hip/hip_runtime.h>

typedef unsigned int u32;
typedef unsigned short u16;

typedef __attribute__((ext_vector_type(16))) __bf16 v16bf;
typedef __attribute__((ext_vector_type(8)))  float  v8f;

struct U4 { u32 x, y, z, w; };
union FragU { U4 q[2]; v16bf v; };

// ---------- helpers ----------
__device__ __forceinline__ u16 f2bf(float f) {
  union { float f; u32 u; } v; v.f = f;
  u32 r = v.u + 0x7FFFu + ((v.u >> 16) & 1u);   // round-to-nearest-even
  return (u16)(r >> 16);
}
__device__ __forceinline__ float sigmoidf_(float x) {
  return 1.0f / (1.0f + __expf(-x));
}

// Problem constants: B=64, T=512, I=256, H=512, 4H=2048
#define NB    64
#define NT    512
#define NI    256
#define NH    512
#define NG    2048

// ---------- one-time prep kernels ----------

// x (B*T*I fp32) -> bf16, 8 elems per thread. grid*block == 8388608/8 exactly.
__global__ void k_convert_x(const float* __restrict__ x, u32* __restrict__ xbf) {
  const int i = blockIdx.x * blockDim.x + threadIdx.x;
  const float* s = x + (size_t)i * 8;
  u32 d[4];
#pragma unroll
  for (int j = 0; j < 4; ++j)
    d[j] = (u32)f2bf(s[2 * j]) | ((u32)f2bf(s[2 * j + 1]) << 16);
  U4 o; o.x = d[0]; o.y = d[1]; o.z = d[2]; o.w = d[3];
  *(U4*)(xbf + (size_t)i * 4) = o;
}

// Pack w[2048 x K] (row-major fp32) into WMMA B-fragment order:
// out[((ntile*ksteps + kstep)*32 + lane)*8 + j] holds the bf16 pair
// (K = kbase+2j, kbase+2j+1) of column n = ntile*16 + lane%16,
// kbase = kstep*32 + (lane/16)*16.  One thread per (ntile,kstep,lane).
__global__ void k_pack_w(const float* __restrict__ w, u32* __restrict__ out, int K) {
  const int idx = blockIdx.x * blockDim.x + threadIdx.x;
  const int ksteps = K >> 5;
  const int lane = idx & 31;
  const int t = idx >> 5;
  const int kstep = t % ksteps;
  const int ntile = t / ksteps;
  if (ntile >= 128) return;
  const int n = ntile * 16 + (lane & 15);
  const int kbase = kstep * 32 + (lane >> 4) * 16;
  const float* src = w + (size_t)n * K + kbase;
  u32 d[8];
#pragma unroll
  for (int j = 0; j < 8; ++j)
    d[j] = (u32)f2bf(src[2 * j]) | ((u32)f2bf(src[2 * j + 1]) << 16);
  u32* o = out + (size_t)idx * 8;
  U4 a; a.x = d[0]; a.y = d[1]; a.z = d[2]; a.w = d[3];
  U4 b; b.x = d[4]; b.y = d[5]; b.z = d[6]; b.w = d[7];
  *(U4*)o = a; *(U4*)(o + 4) = b;
}

// zero c (32768 f32) and h_bf (32768 bf16 = 16384 dwords)
__global__ void k_init(float* __restrict__ c, u32* __restrict__ hbf) {
  const int i = blockIdx.x * blockDim.x + threadIdx.x;
  c[i] = 0.0f;
  if (i < 16384) hbf[i] = 0u;
}

// ---------- phase 1: x_proj = x @ w_ih^T + (b_ih + b_hh), stored [T][B][4H] ----------
// One wave computes 4 M-tiles x 1 N-tile (reuses the B fragment).  This kernel
// has massive TLP (65536 waves) and is HBM-bound (256 MB output stream), so
// per-wave load latency is hidden by occupancy.
// waves = 512 M-groups * 128 ntiles = 65536 -> 8192 blocks of 256.
__global__ void __launch_bounds__(256) k_xproj(const u32* __restrict__ xbf,
                                               const u32* __restrict__ wpack,
                                               const float* __restrict__ b_ih,
                                               const float* __restrict__ b_hh,
                                               float* __restrict__ xproj) {
  const int lane  = threadIdx.x & 31;
  const int wv    = threadIdx.x >> 5;
  const int wid   = blockIdx.x * 8 + wv;
  const int ntile = wid & 127;
  const int mg    = wid >> 7;          // 0..511, each covers 64 rows
  const int sel   = lane >> 4;         // half-wave select
  const int lm    = lane & 15;
  const int n     = ntile * 16 + lm;

  const u16* xb = (const u16*)xbf;
  const u32* bbase = wpack + (((size_t)ntile * 8) * 32 + lane) * 8;

  FragU bbuf[2];
  FragU abuf[2][4];

  // prologue: k-step 0
  {
    const U4* bp = (const U4*)bbase;
    bbuf[0].q[0] = bp[0]; bbuf[0].q[1] = bp[1];
#pragma unroll
    for (int s = 0; s < 4; ++s) {
      const int m = mg * 64 + s * 16 + lm;
      const u16* ap = xb + (size_t)m * NI + sel * 16;
      abuf[0][s].q[0] = *(const U4*)ap;
      abuf[0][s].q[1] = *(const U4*)(ap + 8);
    }
  }

  v8f acc[4] = {};
#pragma unroll
  for (int ks = 0; ks < 8; ++ks) {          // K = 256 = 8 * 32
    const int cur = ks & 1, nxt = cur ^ 1;
    if (ks < 7) {                           // issue next k-step's loads first
      const int k1 = (ks + 1) * 32;
      const U4* bp = (const U4*)(bbase + (size_t)(ks + 1) * 32 * 8);
      bbuf[nxt].q[0] = bp[0]; bbuf[nxt].q[1] = bp[1];
#pragma unroll
      for (int s = 0; s < 4; ++s) {
        const int m = mg * 64 + s * 16 + lm;
        const u16* ap = xb + (size_t)m * NI + k1 + sel * 16;
        abuf[nxt][s].q[0] = *(const U4*)ap;
        abuf[nxt][s].q[1] = *(const U4*)(ap + 8);
      }
    }
#pragma unroll
    for (int s = 0; s < 4; ++s)
      acc[s] = __builtin_amdgcn_wmma_f32_16x16x32_bf16(
          false, abuf[cur][s].v, false, bbuf[cur].v, (short)0, acc[s], false, false);
  }

  const float bias = b_ih[n] + b_hh[n];
#pragma unroll
  for (int s = 0; s < 4; ++s) {
#pragma unroll
    for (int r = 0; r < 8; ++r) {
      const int m = mg * 64 + s * 16 + r + sel * 8;
      const int b = m >> 9;        // / T
      const int t = m & 511;       // % T
      xproj[((size_t)t * NB + b) * NG + n] = acc[s][r] + bias;
    }
  }
}

// ---------- phase 2: one recurrent timestep ----------
// 32 blocks, one per htile.  8 waves = (mtile 0..3) x (kq 0..1): each tile's
// K=512 is split across two waves, combined through LDS.
// The block first stages ALL of its operands in LDS (cooperative bulk
// global_load_b128 -> ds_store_b128, one wait for the whole batch):
//   - the 4 gate-tile weight chunks for this htile (64 KB, fragment order)
//   - the full h state (64 KB bf16)
// so every WMMA fragment read is a cheap ds_load_b128 pair instead of an L2
// round trip -- this kernel is the serial critical path (256 waves only) and
// per-wave latency, not occupancy, determines step time.
// Dynamic LDS: 64K (B) + 64K (h) + 16K (reduction) = 147456 B (<320 KB/WGP).
#define LDS_B_BYTES   65536
#define LDS_H_BYTES   65536
#define LDS_RED_BYTES 16384
#define LDS_STEP_BYTES (LDS_B_BYTES + LDS_H_BYTES + LDS_RED_BYTES)

__global__ void __launch_bounds__(256, 1)
__attribute__((amdgpu_waves_per_eu(1, 2)))
k_step(const u32*  __restrict__ whh_pack,
       const float* __restrict__ xproj,
       const u16*  __restrict__ h_in,
       u16*        __restrict__ h_out,
       float*      __restrict__ c,
       float*      __restrict__ out,
       int t) {
  extern __shared__ char smem[];
  u32*   bsh = (u32*)smem;                         // [g][kstep][lane][8] dwords
  u16*   hsh = (u16*)(smem + LDS_B_BYTES);         // h row-major [64][512]
  float* red = (float*)(smem + LDS_B_BYTES + LDS_H_BYTES); // [mtile][g][256]

  const int tid   = threadIdx.x;
  const int lane  = tid & 31;
  const int wv    = tid >> 5;          // 0..7
  const int htile = blockIdx.x;        // 0..31
  const int mtile = wv & 3;            // 0..3
  const int kq    = wv >> 2;           // K-split half
  const int sel   = lane >> 4;
  const int lm    = lane & 15;

  // ---- cooperative staging: weights (4 gate chunks of 4096 dwords) ----
#pragma unroll
  for (int g = 0; g < 4; ++g) {
    const u32* src = whh_pack + (size_t)(g * 32 + htile) * (16 * 32 * 8);
    u32* dst = bsh + g * 4096;
#pragma unroll
    for (int j = 0; j < 4; ++j) {              // 16 dwords / thread / gate
      const int off = tid * 16 + j * 4;
      *(U4*)(dst + off) = *(const U4*)(src + off);
    }
  }
  // ---- cooperative staging: h state (16384 dwords) ----
  {
    const u32* src = (const u32*)h_in;
    u32* dst = (u32*)hsh;
#pragma unroll
    for (int j = 0; j < 16; ++j) {             // 64 dwords / thread
      const int off = tid * 64 + j * 4;
      *(U4*)(dst + off) = *(const U4*)(src + off);
    }
  }
  __syncthreads();

  // ---- WMMA loop: fragments from LDS, register double-buffered ----
  const u16* ha = hsh + (mtile * 16 + lm) * NH + kq * 256 + sel * 16;
  const u32* bb = bsh + ((kq * 8) * 32 + lane) * 8;

  FragU abuf[2];
  FragU bbuf[2][4];
  abuf[0].q[0] = *(const U4*)ha;
  abuf[0].q[1] = *(const U4*)(ha + 8);
#pragma unroll
  for (int g = 0; g < 4; ++g) {
    const U4* bp = (const U4*)(bb + g * 4096);
    bbuf[0][g].q[0] = bp[0]; bbuf[0][g].q[1] = bp[1];
  }

  v8f acc[4] = {};
#pragma unroll
  for (int ksl = 0; ksl < 8; ++ksl) {          // 8 local k-steps (K range 256)
    const int cur = ksl & 1, nxt = cur ^ 1;
    if (ksl < 7) {                             // issue next k-step's LDS reads
      const u16* ap = ha + (ksl + 1) * 32;
      abuf[nxt].q[0] = *(const U4*)ap;
      abuf[nxt].q[1] = *(const U4*)(ap + 8);
#pragma unroll
      for (int g = 0; g < 4; ++g) {
        const U4* bp = (const U4*)(bb + g * 4096 + (ksl + 1) * 32 * 8);
        bbuf[nxt][g].q[0] = bp[0]; bbuf[nxt][g].q[1] = bp[1];
      }
    }
#pragma unroll
    for (int g = 0; g < 4; ++g)
      acc[g] = __builtin_amdgcn_wmma_f32_16x16x32_bf16(
          false, abuf[cur].v, false, bbuf[cur][g].v, (short)0, acc[g], false, false);
  }

  // ---- cross-wave K reduction through LDS ----
  if (kq == 1) {
#pragma unroll
    for (int g = 0; g < 4; ++g)
      *(v8f*)(red + (mtile * 4 + g) * 256 + lane * 8) = acc[g];
  }
  __syncthreads();
  if (kq == 1) return;

#pragma unroll
  for (int g = 0; g < 4; ++g) {
    const v8f part = *(const v8f*)(red + (mtile * 4 + g) * 256 + lane * 8);
#pragma unroll
    for (int r = 0; r < 8; ++r) acc[g][r] += part[r];
  }

  // ---- fused LSTM cell epilogue ----
  const int n = htile * 16 + lm;                 // hidden index 0..511
  const float* xp_base = xproj + (size_t)t * NB * NG;

  float* outputs = out;                          // [B][T][H]
  float* ht_o = out + (size_t)NB * NT * NH;      // 16777216
  float* ct_o = ht_o + NB * NH;
  float* f_o  = ct_o + NB * NH;
  float* i_o  = f_o  + NB * NH;
  float* o_o  = i_o  + NB * NH;

#pragma unroll
  for (int r = 0; r < 8; ++r) {
    const int m = mtile * 16 + r + sel * 8;      // batch index 0..63
    const float* xp = xp_base + (size_t)m * NG + n;
    const float gf = acc[0][r] + xp[0];
    const float gi = acc[1][r] + xp[512];
    const float gc = acc[2][r] + xp[1024];
    const float go = acc[3][r] + xp[1536];
    const float ft = sigmoidf_(gf);
    const float it = sigmoidf_(gi);
    const float ot = sigmoidf_(go);
    const float cand = __sinf(gc);
    const int ci = m * NH + n;
    const float cn = c[ci] * ft + cand * it;
    c[ci] = cn;
    const float hn = ot * __sinf(cn);
    outputs[(size_t)m * (NT * NH) + (size_t)t * NH + n] = hn;
    h_out[ci] = f2bf(hn);
    if (t == NT - 1) {
      ht_o[ci] = hn; ct_o[ci] = cn; f_o[ci] = ft; i_o[ci] = it; o_o[ci] = ot;
    }
  }
}

// ---------- host ----------
extern "C" void kernel_launch(void* const* d_in, const int* in_sizes, int n_in,
                              void* d_out, int out_size, void* d_ws, size_t ws_size,
                              hipStream_t stream) {
  (void)in_sizes; (void)n_in; (void)out_size; (void)ws_size;
  const float* x    = (const float*)d_in[0];
  const float* w_ih = (const float*)d_in[1];
  const float* w_hh = (const float*)d_in[2];
  const float* b_ih = (const float*)d_in[3];
  const float* b_hh = (const float*)d_in[4];

  char* ws = (char*)d_ws;
  float* xproj =  (float*)(ws);                    // 512*64*2048*4 = 268435456 B
  u32*   xbf   =  (u32*) (ws + 268435456ull);      // 16777216 B
  u32*   wih_p =  (u32*) (ws + 285212672ull);      // 1048576 B
  u32*   whh_p =  (u32*) (ws + 286261248ull);      // 2097152 B
  float* cbuf  =  (float*)(ws + 288358400ull);     // 131072 B
  u16*   hbf0  =  (u16*) (ws + 288489472ull);      // 65536 B
  u16*   hbf1  =  (u16*) (ws + 288555008ull);      // 65536 B

  // one-time prep
  k_convert_x<<<4096, 256, 0, stream>>>(x, xbf);           // 8388608/8 threads
  k_pack_w   <<<128,  256, 0, stream>>>(w_ih, wih_p, NI);  // 128*8*32 threads
  k_pack_w   <<<256,  256, 0, stream>>>(w_hh, whh_p, NH);  // 128*16*32 threads
  k_init     <<<128,  256, 0, stream>>>(cbuf, (u32*)hbf0); // 32768 threads

  // big input-projection GEMM (WMMA bf16 -> fp32)
  k_xproj<<<8192, 256, 0, stream>>>(xbf, wih_p, b_ih, b_hh, xproj);

  // serial recurrence: stream ordering carries the h/c dependency
  for (int t = 0; t < NT; ++t) {
    const u16* hin = (t & 1) ? hbf1 : hbf0;
    u16*       hout = (t & 1) ? hbf0 : hbf1;
    k_step<<<32, 256, LDS_STEP_BYTES, stream>>>(whh_p, xproj, hin, hout, cbuf,
                                                (float*)d_out, t);
  }
}
